// PairStack_63763084476753
// MI455X (gfx1250) — matrix operand order repaired and checked by reading, not compiled, not run on hardware
//
#include <hip/hip_runtime.h>

#define NRES 384
#define NN   (NRES * NRES)
#define CZ   128
#define TM   64          // row-tile per block
#define ASTR 136         // bf16 LDS stride for 128-wide panels (16B-aligned rows, bank-spread)
#define PSTR 132         // f32 LDS stride
#define ESTRA 40         // bf16 LDS stride for 32-wide tiles
#define SSTR 392         // bf16 LDS stride for 384-wide attention rows
#define QTS  64          // attention q rows per block
#define EKB  32          // einsum K-step

typedef __bf16 bf16_t;
typedef bf16_t v16bf __attribute__((ext_vector_type(16)));
typedef float  v8f   __attribute__((ext_vector_type(8)));
typedef unsigned short u16;
typedef u16 u16x8 __attribute__((ext_vector_type(8)));

#define V8F_ZERO {0.f,0.f,0.f,0.f,0.f,0.f,0.f,0.f}

__device__ inline u16 f2bf(float f) {
  unsigned u = __float_as_uint(f);
  return (u16)((u + 0x7FFFu + ((u >> 16) & 1u)) >> 16);
}
__device__ inline float bf2f(u16 h) { return __uint_as_float(((unsigned)h) << 16); }
__device__ inline float sigmoidf_(float x) { return 1.f / (1.f + __expf(-x)); }

// ---- CDNA5 async global->LDS copy (ASYNCcnt-tracked, ISA 08 §4) ----
// VGLOBAL async form: vdst = LDS byte address VGPR, v[addr] = 64-bit global addr.
__device__ inline void async_ld16(unsigned lds_addr, const u16* gaddr) {
  asm volatile("global_load_async_to_lds_b128 %0, %1, off"
               :: "v"(lds_addr), "v"(gaddr) : "memory");
}
__device__ inline void wait_async0() {
  asm volatile("s_wait_asynccnt 0x0" ::: "memory");
}

// ---- WMMA fragment loaders (CDNA5 wave32 16x16x32 bf16 layouts, ISA 7.12.2) ----
__device__ inline v16bf ld_afrag(const u16* A, int stride, int m0, int k0) {
  int lane = threadIdx.x & 31;
  int row  = m0 + (lane & 15);
  int kb   = k0 + ((lane >> 4) << 3);
  const u16* p = A + row * stride + kb;
  union { v16bf v; u16x8 h[2]; } r;
  r.h[0] = *(const u16x8*)(p);
  r.h[1] = *(const u16x8*)(p + 16);
  return r.v;
}
__device__ inline v16bf ld_bfrag(const u16* Bt, int stride, int n0, int k0) {
  int lane = threadIdx.x & 31;
  int row  = n0 + (lane & 15);
  int kk   = k0 + ((lane >> 4) << 3);
  const u16* p = Bt + row * stride + kk;
  union { v16bf v; u16x8 h[2]; } r;
  r.h[0] = *(const u16x8*)(p);
  r.h[1] = *(const u16x8*)(p + 16);
  return r.v;
}
__device__ inline v8f wmma_bf16(v16bf a, v16bf b, v8f c) {
  return __builtin_amdgcn_wmma_f32_16x16x32_bf16(false, a, false, b, (short)0, c, false, false);
}

// 64x128 = [64x128 bf16 A] x [128x128 (Wt stored [o][k])] -> f32 P[64][PSTR]
__device__ inline void gemm_tile_64x128(const u16* A, const u16* Wt, float* P) {
  int wave = threadIdx.x >> 5;
  int lane = threadIdx.x & 31;
  int m0 = (wave & 3) * 16;
  int nbase = (wave >> 2) * 64;
  v16bf a[4];
#pragma unroll
  for (int kk = 0; kk < 4; ++kk) a[kk] = ld_afrag(A, ASTR, m0, kk * 32);
#pragma unroll
  for (int nt = 0; nt < 4; ++nt) {
    int n0 = nbase + nt * 16;
    v8f acc = V8F_ZERO;
#pragma unroll
    for (int kk = 0; kk < 4; ++kk)
      acc = wmma_bf16(a[kk], ld_bfrag(Wt, ASTR, n0, kk * 32), acc);
    int col = n0 + (lane & 15);
    int rb  = m0 + ((lane >> 4) << 3);
#pragma unroll
    for (int v = 0; v < 8; ++v) P[(rb + v) * PSTR + col] = acc[v];
  }
}

// ---- cooperative staging helpers (256 threads) ----
__device__ inline void stage_rows_f32(const float* __restrict__ src, float* P,
                                      int r0, int rowTrans) {
  for (int e = threadIdx.x; e < TM * CZ; e += 256) {
    int rl = e >> 7, c = e & 127;
    int s = r0 + rl;
    int g = rowTrans ? ((s % NRES) * NRES + s / NRES) : s;
    P[rl * PSTR + c] = src[(size_t)g * CZ + c];
  }
}
__device__ inline void ln_to_bf16(const float* P, const float* __restrict__ gam,
                                  const float* __restrict__ bet, u16* dst) {
  int t = threadIdx.x;
  if (t < TM) {
    float m = 0.f;
    for (int c = 0; c < CZ; ++c) m += P[t * PSTR + c];
    m *= (1.f / CZ);
    float v = 0.f;
    for (int c = 0; c < CZ; ++c) { float d = P[t * PSTR + c] - m; v += d * d; }
    v *= (1.f / CZ);
    float rs = rsqrtf(v + 1e-5f);
    for (int c = 0; c < CZ; ++c)
      dst[t * ASTR + c] = f2bf((P[t * PSTR + c] - m) * rs * gam[c] + bet[c]);
  }
}
__device__ inline void stage_wT_cols(const float* __restrict__ W, int ldw, int ocol0, u16* Wt) {
  for (int e = threadIdx.x; e < 128 * CZ; e += 256) {
    int k = e >> 7, o = e & 127;
    Wt[o * ASTR + k] = f2bf(W[(size_t)k * ldw + ocol0 + o]);
  }
}

// ================= Kernel 1: triangle-mul prologue =================
__global__ __launch_bounds__(256) void k_trimul_pre(
    const float* __restrict__ z, const float* __restrict__ lng, const float* __restrict__ lnb,
    const float* __restrict__ apw, const float* __restrict__ apb,
    const float* __restrict__ agw, const float* __restrict__ agb,
    const float* __restrict__ bpw, const float* __restrict__ bpb,
    const float* __restrict__ bgw, const float* __restrict__ bgb,
    const float* __restrict__ gw,  const float* __restrict__ gb,
    u16* __restrict__ aT, u16* __restrict__ bT, float* __restrict__ gbuf)
{
  __shared__ float P[TM * PSTR];
  __shared__ float Q[TM * PSTR];
  __shared__ u16 znb[TM * ASTR];
  __shared__ u16 Wt[CZ * ASTR];
  int r0 = blockIdx.x * TM;

  stage_rows_f32(z, P, r0, 0);
  __syncthreads();
  ln_to_bf16(P, lng, lnb, znb);
  __syncthreads();

  stage_wT_cols(apw, CZ, 0, Wt); __syncthreads();
  gemm_tile_64x128(znb, Wt, P);  __syncthreads();
  stage_wT_cols(agw, CZ, 0, Wt); __syncthreads();
  gemm_tile_64x128(znb, Wt, Q);  __syncthreads();
  for (int e = threadIdx.x; e < TM * CZ; e += 256) {
    int c = e >> 6, rl = e & 63;
    float val = sigmoidf_(Q[rl * PSTR + c] + agb[c]) * (P[rl * PSTR + c] + apb[c]);
    aT[(size_t)c * NN + r0 + rl] = f2bf(val);
  }
  __syncthreads();

  stage_wT_cols(bpw, CZ, 0, Wt); __syncthreads();
  gemm_tile_64x128(znb, Wt, P);  __syncthreads();
  stage_wT_cols(bgw, CZ, 0, Wt); __syncthreads();
  gemm_tile_64x128(znb, Wt, Q);  __syncthreads();
  for (int e = threadIdx.x; e < TM * CZ; e += 256) {
    int c = e >> 6, rl = e & 63;
    float val = sigmoidf_(Q[rl * PSTR + c] + bgb[c]) * (P[rl * PSTR + c] + bpb[c]);
    bT[(size_t)c * NN + r0 + rl] = f2bf(val);
  }
  __syncthreads();

  stage_wT_cols(gw, CZ, 0, Wt); __syncthreads();
  gemm_tile_64x128(znb, Wt, P); __syncthreads();
  for (int e = threadIdx.x; e < TM * CZ; e += 256) {
    int rl = e >> 7, c = e & 127;
    gbuf[(size_t)(r0 + rl) * CZ + c] = sigmoidf_(P[rl * PSTR + c] + gb[c]);
  }
}

// ================= Kernel 2: per-channel triangle einsum =================
// 128x128 output tile / block. TRANS=0: async global->LDS double buffering
// (ASYNCcnt). TRANS=1: register-staged transpose during LDS fill.
template <int TRANS>
__global__ __launch_bounds__(256) void k_tri_einsum(
    const u16* __restrict__ aT, const u16* __restrict__ bT,
    float* __restrict__ xbuf)
{
  __shared__ u16 As[2][128 * ESTRA];
  __shared__ u16 Bs[2][128 * ESTRA];
  int c = blockIdx.z;
  int i0 = blockIdx.x * 128, j0 = blockIdx.y * 128;
  const u16* Ap = aT + (size_t)c * NN;
  const u16* Bp = bT + (size_t)c * NN;
  int tid = threadIdx.x;
  int wave = tid >> 5, lane = tid & 31;
  int m0 = (wave & 3) * 32;   // two m-tiles: m0, m0+16
  int nb = (wave >> 2) * 64;  // four n-tiles: nb + 16*nt

  v8f acc[2][4];
#pragma unroll
  for (int a = 0; a < 2; ++a)
#pragma unroll
    for (int n = 0; n < 4; ++n) acc[a][n] = V8F_ZERO;

  const int NK = NRES / EKB;  // 12

  if (!TRANS) {
    // ---- async-copy double-buffered pipeline ----
    auto issue = [&](int p, int k0) {
#pragma unroll
      for (int j = 0; j < 2; ++j) {
        int q = tid + 256 * j;     // 0..511 16B-chunks per panel
        int row = q >> 2;          // 0..127
        int c8 = (q & 3) << 3;     // k element offset: 0,8,16,24
        unsigned la = (unsigned)(size_t)&As[p][row * ESTRA + c8];
        unsigned lb = (unsigned)(size_t)&Bs[p][row * ESTRA + c8];
        async_ld16(la, Ap + (size_t)(i0 + row) * NRES + k0 + c8);
        async_ld16(lb, Bp + (size_t)(j0 + row) * NRES + k0 + c8);
      }
    };
    issue(0, 0);
    wait_async0();
    __syncthreads();
    for (int ks = 0; ks < NK; ++ks) {
      int p = ks & 1;
      if (ks + 1 < NK) issue(p ^ 1, (ks + 1) * EKB);  // fill other buffer async
      if (ks + 2 < NK && lane < 16) {
        __builtin_prefetch(Ap + (size_t)(i0 + lane * 8) * NRES + (ks + 2) * EKB, 0, 1);
        __builtin_prefetch(Bp + (size_t)(j0 + lane * 8) * NRES + (ks + 2) * EKB, 0, 1);
      }
      v16bf a0 = ld_afrag(As[p], ESTRA, m0, 0);
      v16bf a1 = ld_afrag(As[p], ESTRA, m0 + 16, 0);
#pragma unroll
      for (int nt = 0; nt < 4; ++nt) {
        v16bf bf = ld_bfrag(Bs[p], ESTRA, nb + nt * 16, 0);
        acc[0][nt] = wmma_bf16(a0, bf, acc[0][nt]);
        acc[1][nt] = wmma_bf16(a1, bf, acc[1][nt]);
      }
      wait_async0();
      __syncthreads();
    }
  } else {
    // ---- register-staged transpose pipeline ----
    u16 ta[16], tb[16];
    auto load_step = [&](int k0) {
#pragma unroll
      for (int j = 0; j < 16; ++j) {
        int e = tid + 256 * j;
        int kk = e >> 7;
        int m  = e & 127;
        ta[j] = Ap[(size_t)(k0 + kk) * NRES + i0 + m];
        tb[j] = Bp[(size_t)(k0 + kk) * NRES + j0 + m];
      }
    };
    auto store_step = [&](int p) {
#pragma unroll
      for (int j = 0; j < 16; ++j) {
        int e = tid + 256 * j;
        int kk = e >> 7;
        int m  = e & 127;
        As[p][m * ESTRA + kk] = ta[j];
        Bs[p][m * ESTRA + kk] = tb[j];
      }
    };
    load_step(0);
    store_step(0);
    __syncthreads();
    for (int ks = 0; ks < NK; ++ks) {
      int p = ks & 1;
      if (ks + 1 < NK) load_step((ks + 1) * EKB);
      v16bf a0 = ld_afrag(As[p], ESTRA, m0, 0);
      v16bf a1 = ld_afrag(As[p], ESTRA, m0 + 16, 0);
#pragma unroll
      for (int nt = 0; nt < 4; ++nt) {
        v16bf bf = ld_bfrag(Bs[p], ESTRA, nb + nt * 16, 0);
        acc[0][nt] = wmma_bf16(a0, bf, acc[0][nt]);
        acc[1][nt] = wmma_bf16(a1, bf, acc[1][nt]);
      }
      if (ks + 1 < NK) store_step(p ^ 1);
      __syncthreads();
    }
  }

  int col = lane & 15, rbv = (lane >> 4) << 3;
  float* Xp = xbuf + (size_t)c * NN;
#pragma unroll
  for (int mt = 0; mt < 2; ++mt)
#pragma unroll
    for (int nt = 0; nt < 4; ++nt)
#pragma unroll
      for (int v = 0; v < 8; ++v)
        Xp[(size_t)(i0 + m0 + mt * 16 + rbv + v) * NRES + j0 + nb + nt * 16 + col] =
            acc[mt][nt][v];
}

// ================= Kernel 3: triangle-mul epilogue =================
__global__ __launch_bounds__(256) void k_trimul_post(
    const float* __restrict__ xbuf, const float* __restrict__ gbuf,
    const float* __restrict__ lng, const float* __restrict__ lnb,
    const float* __restrict__ zw, const float* __restrict__ zb,
    float* __restrict__ zbuf)
{
  __shared__ float P[TM * PSTR];
  __shared__ float Q[TM * PSTR];
  __shared__ u16 znb[TM * ASTR];
  __shared__ u16 Wt[CZ * ASTR];
  int r0 = blockIdx.x * TM;
  for (int e = threadIdx.x; e < TM * CZ; e += 256) {
    int c = e >> 6, rl = e & 63;
    P[rl * PSTR + c] = xbuf[(size_t)c * NN + r0 + rl];
  }
  __syncthreads();
  ln_to_bf16(P, lng, lnb, znb);
  __syncthreads();
  stage_wT_cols(zw, CZ, 0, Wt); __syncthreads();
  gemm_tile_64x128(znb, Wt, Q); __syncthreads();
  for (int e = threadIdx.x; e < TM * CZ; e += 256) {
    int rl = e >> 7, c = e & 127;
    size_t gi = (size_t)(r0 + rl) * CZ + c;
    zbuf[gi] += gbuf[gi] * (Q[rl * PSTR + c] + zb[c]);
  }
}

// ================= Kernel 4: attention prologue =================
__global__ __launch_bounds__(256) void k_att_pre(
    const float* __restrict__ z, const float* __restrict__ lng, const float* __restrict__ lnb,
    const float* __restrict__ wq, const float* __restrict__ wk, const float* __restrict__ wv,
    const float* __restrict__ gw, const float* __restrict__ gb,
    const float* __restrict__ biasw,
    u16* __restrict__ qb, u16* __restrict__ kb, u16* __restrict__ vT,
    float* __restrict__ gatt, float* __restrict__ biasbuf, int transposed)
{
  __shared__ float P[TM * PSTR];
  __shared__ u16 znb[TM * ASTR];
  __shared__ u16 Wt[CZ * ASTR];
  int s0 = blockIdx.x * TM;
  int b = s0 / NRES, q0 = s0 % NRES;

  stage_rows_f32(z, P, s0, transposed);
  __syncthreads();
  ln_to_bf16(P, lng, lnb, znb);
  __syncthreads();

  if (threadIdx.x < TM) {
    int t = threadIdx.x;
    int s = s0 + t;
    int g = transposed ? ((s % NRES) * NRES + s / NRES) : s;
    for (int h = 0; h < 4; ++h) {
      float acc = 0.f;
      for (int c = 0; c < CZ; ++c) acc += bf2f(znb[t * ASTR + c]) * biasw[c * 4 + h];
      biasbuf[(size_t)h * NN + g] = acc;
    }
  }
  __syncthreads();

  stage_wT_cols(wq, CZ, 0, Wt); __syncthreads();
  gemm_tile_64x128(znb, Wt, P); __syncthreads();
  for (int e = threadIdx.x; e < TM * CZ; e += 256) {
    int rl = e >> 7, c = e & 127; int h = c >> 5, cc = c & 31;
    qb[((size_t)(b * 4 + h) * NRES + q0 + rl) * 32 + cc] =
        f2bf(P[rl * PSTR + c] * 0.17677669529663687f);
  }
  __syncthreads();
  stage_wT_cols(wk, CZ, 0, Wt); __syncthreads();
  gemm_tile_64x128(znb, Wt, P); __syncthreads();
  for (int e = threadIdx.x; e < TM * CZ; e += 256) {
    int rl = e >> 7, c = e & 127; int h = c >> 5, cc = c & 31;
    kb[((size_t)(b * 4 + h) * NRES + q0 + rl) * 32 + cc] = f2bf(P[rl * PSTR + c]);
  }
  __syncthreads();
  stage_wT_cols(wv, CZ, 0, Wt); __syncthreads();
  gemm_tile_64x128(znb, Wt, P); __syncthreads();
  for (int e = threadIdx.x; e < TM * CZ; e += 256) {
    int c = e >> 6, rl = e & 63; int h = c >> 5, cc = c & 31;
    vT[((size_t)(b * 4 + h) * 32 + cc) * NRES + q0 + rl] = f2bf(P[rl * PSTR + c]);
  }
  __syncthreads();
  stage_wT_cols(gw, CZ, 0, Wt); __syncthreads();
  gemm_tile_64x128(znb, Wt, P); __syncthreads();
  for (int e = threadIdx.x; e < TM * CZ; e += 256) {
    int rl = e >> 7, c = e & 127;
    gatt[(size_t)(s0 + rl) * CZ + c] = sigmoidf_(P[rl * PSTR + c] + gb[c]);
  }
}

// ================= Kernel 5: fused QK^T + softmax + attn*V =================
__global__ __launch_bounds__(256) void k_att_core(
    const u16* __restrict__ qb, const u16* __restrict__ kb, const u16* __restrict__ vT,
    const float* __restrict__ biasbuf, float* __restrict__ obuf, int transposed)
{
  __shared__ u16 As[QTS * SSTR];
  __shared__ u16 Qs[QTS * ESTRA];
  __shared__ u16 Ks[32 * ESTRA];
  __shared__ u16 Vs[32 * ESTRA];
  int q0 = blockIdx.x * QTS;
  int h = blockIdx.y;
  int b = blockIdx.z;
  size_t bh = (size_t)b * 4 + h;
  const u16* Qp = qb + bh * NRES * 32;
  const u16* Kp = kb + bh * NRES * 32;
  const u16* Vp = vT + bh * 32 * NRES;
  int wave = threadIdx.x >> 5, lane = threadIdx.x & 31;
  int m0 = (wave & 3) * 16;

  for (int e = threadIdx.x; e < QTS * 32; e += 256) {
    int r = e >> 5, cc = e & 31;
    Qs[r * ESTRA + cc] = Qp[(size_t)(q0 + r) * 32 + cc];
  }
  int n0s = (wave >> 2) * 16;
  for (int k0 = 0; k0 < NRES; k0 += 32) {
    __syncthreads();
    for (int e = threadIdx.x; e < 32 * 32; e += 256) {
      int r = e >> 5, cc = e & 31;
      Ks[r * ESTRA + cc] = Kp[(size_t)(k0 + r) * 32 + cc];
    }
    __syncthreads();
    v8f acc = V8F_ZERO;
    acc = wmma_bf16(ld_afrag(Qs, ESTRA, m0, 0), ld_bfrag(Ks, ESTRA, n0s, 0), acc);
    int col = lane & 15, rb = m0 + ((lane >> 4) << 3);
#pragma unroll
    for (int v = 0; v < 8; ++v) {
      int qg = q0 + rb + v;
      int kg = k0 + n0s + col;
      float bs = transposed ? biasbuf[(size_t)h * NN + (size_t)kg * NRES + qg]
                            : biasbuf[(size_t)h * NN + (size_t)qg * NRES + kg];
      As[(rb + v) * SSTR + kg] = f2bf(acc[v] + bs);
    }
  }
  __syncthreads();
  if (threadIdx.x < QTS) {
    int r = threadIdx.x;
    float mx = -1e30f;
    for (int k = 0; k < NRES; ++k) mx = fmaxf(mx, bf2f(As[r * SSTR + k]));
    float sum = 0.f;
    for (int k = 0; k < NRES; ++k) {
      float ev = __expf(bf2f(As[r * SSTR + k]) - mx);
      sum += ev;
      As[r * SSTR + k] = f2bf(ev);
    }
    float inv = 1.f / sum;
    for (int k = 0; k < NRES; ++k) As[r * SSTR + k] = f2bf(bf2f(As[r * SSTR + k]) * inv);
  }
  int n0v = (wave >> 2) * 16;
  v8f o = V8F_ZERO;
  for (int k0 = 0; k0 < NRES; k0 += 32) {
    __syncthreads();
    for (int e = threadIdx.x; e < 32 * 32; e += 256) {
      int r = e >> 5, kk = e & 31;
      Vs[r * ESTRA + kk] = Vp[(size_t)r * NRES + k0 + kk];
    }
    __syncthreads();
    o = wmma_bf16(ld_afrag(As, SSTR, m0, k0), ld_bfrag(Vs, ESTRA, n0v, 0), o);
  }
  int col = lane & 15, rb = m0 + ((lane >> 4) << 3);
#pragma unroll
  for (int v = 0; v < 8; ++v)
    obuf[((size_t)b * NRES + q0 + rb + v) * CZ + h * 32 + n0v + col] = o[v];
}

// ================= Kernel 6: attention epilogue =================
__global__ __launch_bounds__(256) void k_att_post(
    const float* __restrict__ obuf, const float* __restrict__ gatt,
    const float* __restrict__ ow, const float* __restrict__ ob,
    float* __restrict__ zbuf, int transposed)
{
  __shared__ float P[TM * PSTR];
  __shared__ u16 znb[TM * ASTR];
  __shared__ u16 Wt[CZ * ASTR];
  int s0 = blockIdx.x * TM;
  for (int e = threadIdx.x; e < TM * CZ; e += 256) {
    int rl = e >> 7, c = e & 127;
    size_t gi = (size_t)(s0 + rl) * CZ + c;
    znb[rl * ASTR + c] = f2bf(gatt[gi] * obuf[gi]);
  }
  __syncthreads();
  stage_wT_cols(ow, CZ, 0, Wt); __syncthreads();
  gemm_tile_64x128(znb, Wt, P); __syncthreads();
  for (int e = threadIdx.x; e < TM * CZ; e += 256) {
    int rl = e >> 7, c = e & 127;
    int s = s0 + rl;
    int g = transposed ? ((s % NRES) * NRES + s / NRES) : s;
    zbuf[(size_t)g * CZ + c] += P[rl * PSTR + c] + ob[c];
  }
}

// ================= Kernel 7: pair transition (128->512->128) =================
__global__ __launch_bounds__(256) void k_transition(
    const float* __restrict__ zin, const float* __restrict__ lng, const float* __restrict__ lnb,
    const float* __restrict__ w1, const float* __restrict__ b1,
    const float* __restrict__ w2, const float* __restrict__ b2,
    float* __restrict__ zbuf)
{
  __shared__ float P[TM * PSTR];
  __shared__ float Q[TM * PSTR];
  __shared__ u16 znb[TM * ASTR];
  __shared__ u16 hb[TM * ASTR];
  __shared__ u16 Wt[CZ * ASTR];
  int r0 = blockIdx.x * TM;
  stage_rows_f32(zin, P, r0, 0);
  __syncthreads();
  ln_to_bf16(P, lng, lnb, znb);
  for (int e = threadIdx.x; e < TM * PSTR; e += 256) Q[e] = 0.f;
  for (int ch = 0; ch < 4; ++ch) {
    __syncthreads();
    stage_wT_cols(w1, 4 * CZ, ch * 128, Wt);
    __syncthreads();
    gemm_tile_64x128(znb, Wt, P);
    __syncthreads();
    for (int e = threadIdx.x; e < TM * CZ; e += 256) {
      int rl = e >> 7, c = e & 127;
      hb[rl * ASTR + c] = f2bf(fmaxf(P[rl * PSTR + c] + b1[ch * 128 + c], 0.f));
    }
    __syncthreads();
    stage_wT_cols(w2 + (size_t)ch * 128 * CZ, CZ, 0, Wt);
    __syncthreads();
    gemm_tile_64x128(hb, Wt, P);
    __syncthreads();
    for (int e = threadIdx.x; e < TM * CZ; e += 256) {
      int rl = e >> 7, c = e & 127;
      Q[rl * PSTR + c] += P[rl * PSTR + c];
    }
  }
  __syncthreads();
  for (int e = threadIdx.x; e < TM * CZ; e += 256) {
    int rl = e >> 7, c = e & 127;
    zbuf[(size_t)(r0 + rl) * CZ + c] += Q[rl * PSTR + c] + b2[c];
  }
}

// =========================== host launcher ===========================
extern "C" void kernel_launch(void* const* d_in, const int* in_sizes, int n_in,
                              void* d_out, int out_size, void* d_ws, size_t ws_size,
                              hipStream_t stream) {
  (void)in_sizes; (void)n_in; (void)out_size; (void)ws_size;
  auto F = [&](int i) { return (const float*)d_in[i]; };

  char* w = (char*)d_ws;
  float* zbuf = (float*)w;   w += (size_t)NN * CZ * 4;
  u16*   aT   = (u16*)w;     w += (size_t)NN * CZ * 2;   // reused as qb
  u16*   bT   = (u16*)w;     w += (size_t)NN * CZ * 2;   // reused as kb
  float* gbuf = (float*)w;   w += (size_t)NN * CZ * 4;   // reused as gatt
  float* xbuf = (float*)w;   w += (size_t)NN * CZ * 4;   // reused as obuf
  u16*   vT   = (u16*)w;     w += (size_t)NN * CZ * 2;
  float* biasbuf = (float*)w; w += (size_t)4 * NN * 4;

  dim3 blk(256);
  int rowBlocks = NN / TM;

  hipMemcpyAsync(zbuf, d_in[0], (size_t)NN * CZ * 4, hipMemcpyDeviceToDevice, stream);

  // ---- triangle multiplication, outgoing (params 1..16) ----
  k_trimul_pre<<<rowBlocks, blk, 0, stream>>>(zbuf, F(1), F(2), F(5), F(6), F(7), F(8),
                                              F(9), F(10), F(11), F(12), F(13), F(14),
                                              aT, bT, gbuf);
  k_tri_einsum<0><<<dim3(3, 3, 128), blk, 0, stream>>>(aT, bT, xbuf);
  k_trimul_post<<<rowBlocks, blk, 0, stream>>>(xbuf, gbuf, F(3), F(4), F(15), F(16), zbuf);

  // ---- triangle multiplication, incoming (params 17..32) ----
  k_trimul_pre<<<rowBlocks, blk, 0, stream>>>(zbuf, F(17), F(18), F(21), F(22), F(23), F(24),
                                              F(25), F(26), F(27), F(28), F(29), F(30),
                                              aT, bT, gbuf);
  k_tri_einsum<1><<<dim3(3, 3, 128), blk, 0, stream>>>(aT, bT, xbuf);
  k_trimul_post<<<rowBlocks, blk, 0, stream>>>(xbuf, gbuf, F(19), F(20), F(31), F(32), zbuf);

  // ---- triangle attention, starting node (params 33..42) ----
  k_att_pre<<<rowBlocks, blk, 0, stream>>>(zbuf, F(33), F(34), F(36), F(37), F(38),
                                           F(39), F(40), F(35),
                                           aT, bT, vT, gbuf, biasbuf, 0);
  k_att_core<<<dim3(6, 4, 384), blk, 0, stream>>>(aT, bT, vT, biasbuf, xbuf, 0);
  k_att_post<<<rowBlocks, blk, 0, stream>>>(xbuf, gbuf, F(41), F(42), zbuf, 0);

  // ---- triangle attention, ending node (params 43..52) ----
  k_att_pre<<<rowBlocks, blk, 0, stream>>>(zbuf, F(43), F(44), F(46), F(47), F(48),
                                           F(49), F(50), F(45),
                                           aT, bT, vT, gbuf, biasbuf, 1);
  k_att_core<<<dim3(6, 4, 384), blk, 0, stream>>>(aT, bT, vT, biasbuf, xbuf, 1);
  k_att_post<<<rowBlocks, blk, 0, stream>>>(xbuf, gbuf, F(51), F(52), zbuf, 1);

  // ---- pair transition (params 53..58) ----
  k_transition<<<rowBlocks, blk, 0, stream>>>(zbuf, F(53), F(54), F(55), F(56), F(57), F(58),
                                              zbuf);

  hipMemcpyAsync(d_out, zbuf, (size_t)NN * CZ * 4, hipMemcpyDeviceToDevice, stream);
}
